// TasteNet_45483703665013
// MI455X (gfx1250) — matrix operand order repaired and checked by reading, not compiled, not run on hardware
//
#include <hip/hip_runtime.h>
#include <hip/hip_bf16.h>

typedef __attribute__((ext_vector_type(16))) _Float16 v16h;
typedef __attribute__((ext_vector_type(8)))  _Float16 v8h;
typedef __attribute__((ext_vector_type(8)))  float    v8f;

#define NROWS 524288
#define KZ 64            // NUM_SD
#define HD 128           // hidden dim
#define F_REAL 12        // 9 taste + 3 intercept
#define F_PAD 16         // padded output cols for one 16x16 WMMA tile
#define WAVES 4
#define BLOCK_T (WAVES * 32)
#define ROWS_PER_WAVE 16
#define ROWS_PER_BLOCK (WAVES * ROWS_PER_WAVE)

#define WMMA_F16(A, B, C) \
    __builtin_amdgcn_wmma_f32_16x16x32_f16(false, (A), false, (B), (short)0, (C), false, false)

__global__ __launch_bounds__(BLOCK_T) void tastenet_wmma_kernel(
    const float* __restrict__ x,  const float* __restrict__ z,
    const float* __restrict__ W1, const float* __restrict__ b1,
    const float* __restrict__ W2, const float* __restrict__ b2,
    const float* __restrict__ W3, const float* __restrict__ b3,
    float* __restrict__ out)
{
    // smemA holds W1^T (f16, column-major) during layer 1, then becomes the
    // per-wave fp16 activation scratch (4 waves x 16x128 halves = 8192 halves).
    __shared__ _Float16 smemA[KZ * HD];        // 16 KB
    __shared__ _Float16 W2t[HD * HD];          // 32 KB, column-major (transposed)
    __shared__ _Float16 W3t[F_PAD * HD];       // 4 KB, column-major, cols >= 12 zero
    __shared__ float b1s[HD], b2s[HD], b3s[F_PAD];
    __shared__ float obuf[WAVES][F_PAD * 16];  // 16x16 f32 result tile per wave

    const int tid   = threadIdx.x;
    const int wave  = tid >> 5;
    const int lane  = tid & 31;
    const int nlane = lane & 15;
    const int hihalf = lane >> 4;      // 0: lanes 0-15, 1: lanes 16-31
    const int koff  = hihalf * 16;     // B-fragment K offset within a 32-K step
    const int aoff  = hihalf * 16;     // A-fragment K offset within a 32-K step

    // ---------------- stage weights: f32 -> f16, transposed for vector LDS reads
    for (int i = tid; i < KZ * HD; i += BLOCK_T) {
        int r = i / HD, c = i % HD;            // W1 is [64][128] row-major
        smemA[c * KZ + r] = (_Float16)W1[i];
    }
    for (int i = tid; i < HD * HD; i += BLOCK_T) {
        int r = i >> 7, c = i & (HD - 1);      // W2 is [128][128] row-major
        W2t[c * HD + r] = (_Float16)W2[i];
    }
    for (int i = tid; i < F_PAD * HD; i += BLOCK_T) {
        int c = i >> 7, r = i & (HD - 1);      // W3 is [128][12] row-major
        W3t[i] = (c < F_REAL) ? (_Float16)W3[r * F_REAL + c] : (_Float16)0.0f;
    }
    if (tid < HD)    { b1s[tid] = b1[tid]; b2s[tid] = b2[tid]; }
    if (tid < F_PAD) { b3s[tid] = (tid < F_REAL) ? b3[tid] : 0.0f; }
    __syncthreads();

    const int rowBase = blockIdx.x * ROWS_PER_BLOCK + wave * ROWS_PER_WAVE;
    const int mrow    = nlane;                 // this lane's A-matrix row

    // ---------------- build A fragments of z (16 rows x 64 K), f32 -> f16
    // contiguous k order used here (B reads match): element i of the per-lane v16h holds
    //   K = s*32 + (lane>=16 ? 16 : 0) + i
    const float* zr = z + (size_t)(rowBase + mrow) * KZ;
    v16h az[2];
    #pragma unroll
    for (int s = 0; s < 2; ++s) {
        #pragma unroll
        for (int g = 0; g < 2; ++g) {
            const float4* p = (const float4*)(zr + s * 32 + g * 8 + aoff);
            float4 q0 = p[0], q1 = p[1];
            az[s][g * 8 + 0] = (_Float16)q0.x; az[s][g * 8 + 1] = (_Float16)q0.y;
            az[s][g * 8 + 2] = (_Float16)q0.z; az[s][g * 8 + 3] = (_Float16)q0.w;
            az[s][g * 8 + 4] = (_Float16)q1.x; az[s][g * 8 + 5] = (_Float16)q1.y;
            az[s][g * 8 + 6] = (_Float16)q1.z; az[s][g * 8 + 7] = (_Float16)q1.w;
        }
    }

    // ---------------- layer 1: h1 = relu(z @ W1 + b1), K = 64 (2 steps of 32)
    v8f acc[8];
    #pragma unroll
    for (int nt = 0; nt < 8; ++nt)
        #pragma unroll
        for (int r = 0; r < 8; ++r) acc[nt][r] = 0.0f;

    #pragma unroll
    for (int nt = 0; nt < 8; ++nt) {
        const _Float16* wcol = smemA + (nt * 16 + nlane) * KZ;
        v16h bf0 = *(const v16h*)(wcol + 0  + koff);
        v16h bf1 = *(const v16h*)(wcol + 32 + koff);
        acc[nt] = WMMA_F16(az[0], bf0, acc[nt]);
        acc[nt] = WMMA_F16(az[1], bf1, acc[nt]);
    }

    __syncthreads();   // all waves done reading W1^T; smemA becomes h scratch

    _Float16* hb = smemA + wave * (ROWS_PER_WAVE * HD);   // 2048 halves / wave

    // bias + relu, store h1 fp16 row-major [16][128] from the C/D layout:
    // lane<16: rows 0..7 (vgpr r), col = nt*16+lane; lane>=16: rows 8..15
    #pragma unroll
    for (int nt = 0; nt < 8; ++nt) {
        float bias = b1s[nt * 16 + nlane];
        #pragma unroll
        for (int r = 0; r < 8; ++r) {
            float v = acc[nt][r] + bias;
            v = v > 0.0f ? v : 0.0f;
            hb[(hihalf * 8 + r) * HD + nt * 16 + nlane] = (_Float16)v;
        }
    }

    // ---------------- layer 2: h2 = relu(h1 @ W2 + b2), K = 128 (4 steps)
    v8f acc2[8];
    #pragma unroll
    for (int nt = 0; nt < 8; ++nt)
        #pragma unroll
        for (int r = 0; r < 8; ++r) acc2[nt][r] = 0.0f;

    #pragma unroll
    for (int s = 0; s < 4; ++s) {
        v8h lo = *(const v8h*)(hb + mrow * HD + s * 32 + aoff);
        v8h hi = *(const v8h*)(hb + mrow * HD + s * 32 + 8 + aoff);
        v16h a = __builtin_shufflevector(lo, hi, 0,1,2,3,4,5,6,7,8,9,10,11,12,13,14,15);
        #pragma unroll
        for (int nt = 0; nt < 8; ++nt) {
            v16h bf = *(const v16h*)(W2t + (nt * 16 + nlane) * HD + s * 32 + koff);
            acc2[nt] = WMMA_F16(a, bf, acc2[nt]);
        }
    }

    // bias + relu, overwrite scratch with h2 (all layer-2 A-frags consumed)
    #pragma unroll
    for (int nt = 0; nt < 8; ++nt) {
        float bias = b2s[nt * 16 + nlane];
        #pragma unroll
        for (int r = 0; r < 8; ++r) {
            float v = acc2[nt][r] + bias;
            v = v > 0.0f ? v : 0.0f;
            hb[(hihalf * 8 + r) * HD + nt * 16 + nlane] = (_Float16)v;
        }
    }

    // ---------------- layer 3: b = h2 @ W3 + b3 (one 16-col tile, cols>=12 pad)
    v8f acc3;
    #pragma unroll
    for (int r = 0; r < 8; ++r) acc3[r] = 0.0f;

    #pragma unroll
    for (int s = 0; s < 4; ++s) {
        v8h lo = *(const v8h*)(hb + mrow * HD + s * 32 + aoff);
        v8h hi = *(const v8h*)(hb + mrow * HD + s * 32 + 8 + aoff);
        v16h a = __builtin_shufflevector(lo, hi, 0,1,2,3,4,5,6,7,8,9,10,11,12,13,14,15);
        v16h bf = *(const v16h*)(W3t + nlane * HD + s * 32 + koff);
        acc3 = WMMA_F16(a, bf, acc3);
    }

    {
        float bias = b3s[nlane];
        #pragma unroll
        for (int r = 0; r < 8; ++r)
            obuf[wave][(hihalf * 8 + r) * 16 + nlane] = acc3[r] + bias;
    }

    // make this wave's obuf stores visible before per-row reads
    asm volatile("s_wait_dscnt 0" ::: "memory");

    // ---------------- epilogue: clamps + segment dot products (1 lane per row)
    if (lane < 16) {
        const float* ob = obuf[wave] + lane * 16;
        float t[F_REAL];
        #pragma unroll
        for (int j = 0; j < F_REAL; ++j) t[j] = ob[j];
        #pragma unroll
        for (int j = 0; j < 6; ++j) t[j] = t[j] < 0.0f ? t[j] : 0.0f;  // -relu(-b)
        t[7] = t[7] < 0.0f ? t[7] : 0.0f;
        t[8] = t[8] < 0.0f ? t[8] : 0.0f;

        const size_t row = (size_t)rowBase + lane;
        const float* xr = x + row * 9;
        float l0 = xr[0] * t[0] + xr[1] * t[1] + xr[2] * t[2] + t[9];
        float l1 = xr[3] * t[3] + xr[4] * t[4] + xr[5] * t[5] + t[10];
        float l2 = xr[6] * t[6] + xr[7] * t[7] + xr[8] * t[8] + t[11];
        out[row * 3 + 0] = l0;
        out[row * 3 + 1] = l1;
        out[row * 3 + 2] = l2;
    }
}

extern "C" void kernel_launch(void* const* d_in, const int* in_sizes, int n_in,
                              void* d_out, int out_size, void* d_ws, size_t ws_size,
                              hipStream_t stream) {
    const float* x  = (const float*)d_in[0];
    const float* z  = (const float*)d_in[1];
    const float* W1 = (const float*)d_in[2];
    const float* b1 = (const float*)d_in[3];
    const float* W2 = (const float*)d_in[4];
    const float* b2 = (const float*)d_in[5];
    const float* W3 = (const float*)d_in[6];
    const float* b3 = (const float*)d_in[7];
    float* out = (float*)d_out;

    dim3 grid(NROWS / ROWS_PER_BLOCK);
    dim3 block(BLOCK_T);
    hipLaunchKernelGGL(tastenet_wmma_kernel, grid, block, 0, stream,
                       x, z, W1, b1, W2, b2, W3, b3, out);
}
